// Model_20641612825393
// MI455X (gfx1250) — matrix operand order repaired and checked by reading, not compiled, not run on hardware
//
#include <hip/hip_runtime.h>

// ---------------------------------------------------------------------------
// Fused per-species MLP (3x GEMM + silu + dot + segment-sum) for gfx1250.
// bf16 WMMA (v_wmma_f32_16x16x32_bf16), f32 accumulate, f32 atomics for the
// segment reduction. 128-atom M tiles halve L2 weight traffic. Staging is
// split into load-to-register / store-to-LDS phases around the wmma block so
// HBM latency for slice s+1 hides under slice s's 16 wmma ops.
// ---------------------------------------------------------------------------

typedef __attribute__((ext_vector_type(16))) __bf16 v16bf;
typedef __attribute__((ext_vector_type(8)))  __bf16 v8bf;   // 16 bytes
typedef __attribute__((ext_vector_type(4)))  __bf16 v4bf;   // 8 bytes
typedef __attribute__((ext_vector_type(2)))  __bf16 v2bf;   // 4 bytes
typedef __attribute__((ext_vector_type(8)))  float  v8f;

#define N_ATOMS_C   65536
#define N_FEAT_C    2048
#define HIDDEN_C    256
#define N_SPECIES_C 4
#define ATOMS_WG    128           // atoms per workgroup (M tile)
#define NWAVES      8             // wave32 x 8 = 256 threads
#define NTHREADS    (NWAVES * 32)
#define NTJ         8             // N-tiles per wave (half of HIDDEN/16)

__device__ __forceinline__ float silu_f(float x) {
    return x / (1.0f + __expf(-x));
}

// Ping-pong staging buffers + activation buffers shared across waves (~179 KB).
struct __align__(64) Smem {
    __bf16 X[2][ATOMS_WG * 32];        // 16 KB : feature K-slice, atom-major, stride 32
    __bf16 W[2][HIDDEN_C * 32];        // 32 KB : weight K-slice, N-major (transposed)
    __bf16 H1[ATOMS_WG * HIDDEN_C];    // 64 KB : activations, stride 256
    __bf16 H2[ATOMS_WG * HIDDEN_C];    // 64 KB
    float  bias[3][HIDDEN_C];          // 3 KB  : b1/b2/b3 staged once
};

// A fragment, 16x32 bf16 (ISA 7.12.2): lane half h -> VGPR0-3 = K[h*8..h*8+7],
// VGPR4-7 = K[16+h*8 .. 16+h*8+7]. Two 16B LDS loads.
__device__ __forceinline__ v16bf load_afrag(const __bf16* rowbase, int half) {
    union { v16bf v; struct { v8bf lo, hi; } p; } u;
    u.p.lo = *(const v8bf*)(rowbase + half * 8);
    u.p.hi = *(const v8bf*)(rowbase + 16 + half * 8);
    return u.v;
}

// B fragment, 32x16 bf16: same K order as A: lane half h -> K[h*8..h*8+7],
// then K[16+h*8 .. 16+h*8+7]. Two 16B LDS loads from the N-major slice.
__device__ __forceinline__ v16bf load_bfrag(const __bf16* rowbase, int half) {
    union { v16bf v; struct { v8bf lo, hi; } p; } u;
    u.p.lo = *(const v8bf*)(rowbase + half * 8);
    u.p.hi = *(const v8bf*)(rowbase + 16 + half * 8);
    return u.v;
}

// ---- Split staging: load fp32 into registers, later convert+store to LDS ---

// X slice: ATOMS_WG x 32 fp32, 4 float4 per thread.
__device__ __forceinline__ void loadX_regs(const float* __restrict__ feat,
                                           int atom0, int k0, int tid,
                                           float4 xr[4]) {
    #pragma unroll
    for (int u = 0; u < 4; ++u) {
        const int i  = tid + u * NTHREADS;
        const int a  = i >> 3;
        const int k4 = (i & 7) << 2;
        xr[u] = *(const float4*)(feat + (size_t)(atom0 + a) * N_FEAT_C + k0 + k4);
    }
}
__device__ __forceinline__ void storeX_regs(__bf16* sX, int tid,
                                            const float4 xr[4]) {
    #pragma unroll
    for (int u = 0; u < 4; ++u) {
        const int i  = tid + u * NTHREADS;
        const int a  = i >> 3;
        const int k4 = (i & 7) << 2;
        v4bf p;
        p[0] = (__bf16)xr[u].x; p[1] = (__bf16)xr[u].y;
        p[2] = (__bf16)xr[u].z; p[3] = (__bf16)xr[u].w;
        *(v4bf*)(sX + a * 32 + k4) = p;
    }
}

// W slice: 32x256 fp32 (row-major k*256+n) -> LDS transposed sW[n*32+k].
// Two consecutive k-rows per unit so each column stores a packed (k,k+1)
// bf16 pair as one aligned b32 DS store. 2 float4 loads x 4 units per thread.
__device__ __forceinline__ void loadW_regs(const float* __restrict__ W,
                                           int k0, int tid,
                                           float4 wr0[4], float4 wr1[4]) {
    #pragma unroll
    for (int u = 0; u < 4; ++u) {
        const int i  = tid + u * NTHREADS;
        const int k2 = (i >> 6) << 1;
        const int n4 = (i & 63) << 2;
        wr0[u] = *(const float4*)(W + (size_t)(k0 + k2) * HIDDEN_C + n4);
        wr1[u] = *(const float4*)(W + (size_t)(k0 + k2 + 1) * HIDDEN_C + n4);
    }
}
__device__ __forceinline__ void storeW_regs(__bf16* sW, int tid,
                                            const float4 wr0[4], const float4 wr1[4]) {
    #pragma unroll
    for (int u = 0; u < 4; ++u) {
        const int i  = tid + u * NTHREADS;
        const int k2 = (i >> 6) << 1;
        const int n4 = (i & 63) << 2;
        v2bf p;
        p[0] = (__bf16)wr0[u].x; p[1] = (__bf16)wr1[u].x; *(v2bf*)(sW + (n4 + 0) * 32 + k2) = p;
        p[0] = (__bf16)wr0[u].y; p[1] = (__bf16)wr1[u].y; *(v2bf*)(sW + (n4 + 1) * 32 + k2) = p;
        p[0] = (__bf16)wr0[u].z; p[1] = (__bf16)wr1[u].z; *(v2bf*)(sW + (n4 + 2) * 32 + k2) = p;
        p[0] = (__bf16)wr0[u].w; p[1] = (__bf16)wr1[u].w; *(v2bf*)(sW + (n4 + 3) * 32 + k2) = p;
    }
}

// 16 wmma for one K-slice: 2 A frags (32 atoms) x 8 B frags (128 cols).
// Each B fragment feeds two wmma ops; co-resident waves cover the DS waits.
__device__ __forceinline__ void kslice_mma(v8f acc[2][NTJ],
                                           const __bf16* aBase0, int aStride,
                                           const __bf16* sW, int nh,
                                           int lrow, int half) {
    const v16bf a0 = load_afrag(aBase0, half);
    const v16bf a1 = load_afrag(aBase0 + 16 * aStride, half);
    #pragma unroll
    for (int j = 0; j < NTJ; ++j) {
        const v16bf b = load_bfrag(sW + ((nh * NTJ + j) * 16 + lrow) * 32, half);
        acc[0][j] = __builtin_amdgcn_wmma_f32_16x16x32_bf16(
                        false, a0, false, b, (short)0, acc[0][j], false, false);
        acc[1][j] = __builtin_amdgcn_wmma_f32_16x16x32_bf16(
                        false, a1, false, b, (short)0, acc[1][j], false, false);
    }
}

// C/D 16x16 f32 layout: lane n = L&15, VGPR r holds M = (L>>4)*8 + r.
// Apply bias + silu, store bf16 into activation buffer (stride 256).
__device__ __forceinline__ void store_h(const v8f acc[2][NTJ], __bf16* outH,
                                        const float* sbias,
                                        int mb, int nh, int lrow, int half) {
    #pragma unroll
    for (int blk = 0; blk < 2; ++blk) {
        #pragma unroll
        for (int j = 0; j < NTJ; ++j) {
            const int n = (nh * NTJ + j) * 16 + lrow;
            const float b = sbias[n];
            #pragma unroll
            for (int r = 0; r < 8; ++r) {
                const float v = acc[blk][j][r] + b;
                outH[(mb * 32 + blk * 16 + half * 8 + r) * HIDDEN_C + n] =
                    (__bf16)silu_f(v);
            }
        }
    }
}

__global__ __launch_bounds__(NTHREADS, 1)
void fused_species_mlp(const float* __restrict__ feat,
                       const float* __restrict__ W1, const float* __restrict__ B1,
                       const float* __restrict__ W2, const float* __restrict__ B2,
                       const float* __restrict__ W3, const float* __restrict__ B3,
                       const float* __restrict__ W4, const float* __restrict__ B4,
                       const int*   __restrict__ sidx,
                       float* __restrict__ out) {
    __shared__ Smem sm;

    const int tid  = threadIdx.x;
    const int lane = tid & 31;
    const int wave = tid >> 5;
    const int lrow = lane & 15;
    const int half = lane >> 4;
    const int mb   = wave & 3;        // atom quarter (32 atoms)
    const int nh   = wave >> 2;       // N half (128 cols)

    const int atom0 = blockIdx.x * ATOMS_WG;
    const int sp    = atom0 / (N_ATOMS_C / N_SPECIES_C);   // 128 | 16384 -> uniform

    const float* W1s = W1 + (size_t)sp * N_FEAT_C * HIDDEN_C;
    const float* W2s = W2 + (size_t)sp * HIDDEN_C * HIDDEN_C;
    const float* W3s = W3 + (size_t)sp * HIDDEN_C * HIDDEN_C;
    const float* W4s = W4 + (size_t)sp * HIDDEN_C;

    v8f acc[2][NTJ];
    const v8f vzero = {0.f, 0.f, 0.f, 0.f, 0.f, 0.f, 0.f, 0.f};
    float4 xr[4], wr0[4], wr1[4];

    // Prologue: biases (once) + slice 0 of layer 1 staged directly.
    for (int i = tid; i < 3 * HIDDEN_C; i += NTHREADS) {
        const int l = i >> 8, n = i & 255;
        const float* bsrc = (l == 0) ? B1 : (l == 1) ? B2 : B3;
        sm.bias[l][n] = bsrc[sp * HIDDEN_C + n];
    }
    loadX_regs(feat, atom0, 0, tid, xr);  storeX_regs(sm.X[0], tid, xr);
    loadW_regs(W1s, 0, tid, wr0, wr1);    storeW_regs(sm.W[0], tid, wr0, wr1);

    // ------------------------- Layer 1: X @ W1 --------------------------
    #pragma unroll
    for (int blk = 0; blk < 2; ++blk)
        #pragma unroll
        for (int j = 0; j < NTJ; ++j) acc[blk][j] = vzero;

    for (int s = 0; s < N_FEAT_C / 32; ++s) {
        const int cur = s & 1;
        const bool more = (s + 1 < N_FEAT_C / 32);
        __syncthreads();                 // slice s staged; alt buffer free
        if (more) {                      // issue global loads for slice s+1
            loadX_regs(feat, atom0, (s + 1) * 32, tid, xr);
            loadW_regs(W1s, (s + 1) * 32, tid, wr0, wr1);
            if (s + 2 < N_FEAT_C / 32)   // global_prefetch_b8, 2 slices ahead
                __builtin_prefetch(feat + (size_t)(atom0 + (tid >> 3)) * N_FEAT_C
                                        + (s + 2) * 32 + ((tid & 7) << 2), 0, 3);
        }
        kslice_mma(acc, sm.X[cur] + (mb * 32 + lrow) * 32, 32,
                   sm.W[cur], nh, lrow, half);   // loads in flight under wmma
        if (more) {                      // drain loads, convert, store to alt
            storeX_regs(sm.X[cur ^ 1], tid, xr);
            storeW_regs(sm.W[cur ^ 1], tid, wr0, wr1);
        }
    }
    store_h(acc, sm.H1, sm.bias[0], mb, nh, lrow, half);  // H rows shared by waves mb, mb+4

    // ------------------------- Layer 2: H1 @ W2 -------------------------
    #pragma unroll
    for (int blk = 0; blk < 2; ++blk)
        #pragma unroll
        for (int j = 0; j < NTJ; ++j) acc[blk][j] = vzero;

    __syncthreads();                     // last layer-1 reads done
    loadW_regs(W2s, 0, tid, wr0, wr1);   storeW_regs(sm.W[0], tid, wr0, wr1);
    for (int s = 0; s < HIDDEN_C / 32; ++s) {
        const int cur = s & 1;
        const bool more = (s + 1 < HIDDEN_C / 32);
        __syncthreads();
        if (more) loadW_regs(W2s, (s + 1) * 32, tid, wr0, wr1);
        kslice_mma(acc, sm.H1 + (mb * 32 + lrow) * HIDDEN_C + s * 32, HIDDEN_C,
                   sm.W[cur], nh, lrow, half);
        if (more) storeW_regs(sm.W[cur ^ 1], tid, wr0, wr1);
    }
    store_h(acc, sm.H2, sm.bias[1], mb, nh, lrow, half);

    // ------------------------- Layer 3: H2 @ W3 -------------------------
    #pragma unroll
    for (int blk = 0; blk < 2; ++blk)
        #pragma unroll
        for (int j = 0; j < NTJ; ++j) acc[blk][j] = vzero;

    __syncthreads();
    loadW_regs(W3s, 0, tid, wr0, wr1);   storeW_regs(sm.W[0], tid, wr0, wr1);
    for (int s = 0; s < HIDDEN_C / 32; ++s) {
        const int cur = s & 1;
        const bool more = (s + 1 < HIDDEN_C / 32);
        __syncthreads();
        if (more) loadW_regs(W3s, (s + 1) * 32, tid, wr0, wr1);
        kslice_mma(acc, sm.H2 + (mb * 32 + lrow) * HIDDEN_C + s * 32, HIDDEN_C,
                   sm.W[cur], nh, lrow, half);
        if (more) storeW_regs(sm.W[cur ^ 1], tid, wr0, wr1);
    }
    store_h(acc, sm.H1, sm.bias[2], mb, nh, lrow, half);  // reuse H1 for h3

    // -------------- Layer 4: e = h3 . W4 + b4 ; segment sum -------------
    __syncthreads();                     // h3 read cross-wave below
    if (tid < ATOMS_WG) {
        const __bf16* h = sm.H1 + tid * HIDDEN_C;
        float e = 0.f;
        #pragma unroll 8
        for (int n = 0; n < HIDDEN_C; ++n)
            e += (float)h[n] * W4s[n];
        e += B4[sp];
        atomicAdd(out + sidx[atom0 + tid], e);
    }
}

__global__ void zero_out_kernel(float* __restrict__ out, int n) {
    const int i = blockIdx.x * blockDim.x + threadIdx.x;
    if (i < n) out[i] = 0.f;
}

extern "C" void kernel_launch(void* const* d_in, const int* in_sizes, int n_in,
                              void* d_out, int out_size, void* d_ws, size_t ws_size,
                              hipStream_t stream) {
    (void)in_sizes; (void)n_in; (void)d_ws; (void)ws_size;
    const float* feat = (const float*)d_in[0];
    const float* W1   = (const float*)d_in[1];
    const float* B1   = (const float*)d_in[2];
    const float* W2   = (const float*)d_in[3];
    const float* B2   = (const float*)d_in[4];
    const float* W3   = (const float*)d_in[5];
    const float* B3   = (const float*)d_in[6];
    const float* W4   = (const float*)d_in[7];
    const float* B4   = (const float*)d_in[8];
    const int*   sidx = (const int*)d_in[9];
    float* out = (float*)d_out;

    zero_out_kernel<<<(out_size + 255) / 256, 256, 0, stream>>>(out, out_size);
    fused_species_mlp<<<N_ATOMS_C / ATOMS_WG, NTHREADS, 0, stream>>>(
        feat, W1, B1, W2, B2, W3, B3, W4, B4, sidx, out);
}
